// SmoothedMetricLoss_3238405341329
// MI455X (gfx1250) — matrix operand order repaired and checked by reading, not compiled, not run on hardware
//
#include <hip/hip_runtime.h>
#include <hip/hip_bf16.h>

typedef __attribute__((ext_vector_type(16))) _Float16 v16h;
typedef __attribute__((ext_vector_type(8)))  _Float16 v8h;
typedef __attribute__((ext_vector_type(8)))  float    v8f;

#define EMB_D      128
#define B_ROWS     8192
#define N_TILES    (B_ROWS / 16)
#define N_PAIRS    (B_ROWS / 2)
#define MARGIN     1.0f
#define EPS_LOSS   1e-8f

#define LOG2E      1.44269504088896340736f
#define K1_EXP     (LOG2E / (float)EMB_D)   // exp(c/d + m) = exp2(c*K1 + K2)
#define K2_EXP     (LOG2E * MARGIN)

// ---------------------------------------------------------------------------
// Kernel 1: convert fp32 X -> f16 Xh (row-major 8192x128).
// ---------------------------------------------------------------------------
__global__ void cvt_f32_to_f16(const float* __restrict__ X,
                               _Float16* __restrict__ Xh, int n) {
  const int i = (blockIdx.x * blockDim.x + threadIdx.x) * 8;
  if (i + 7 < n) {
    const float4 a = *(const float4*)(X + i);
    const float4 b = *(const float4*)(X + i + 4);
    v8h h;
    h[0] = (_Float16)a.x; h[1] = (_Float16)a.y;
    h[2] = (_Float16)a.z; h[3] = (_Float16)a.w;
    h[4] = (_Float16)b.x; h[5] = (_Float16)b.y;
    h[6] = (_Float16)b.z; h[7] = (_Float16)b.w;
    *(v8h*)(Xh + i) = h;
  }
}

// ---------------------------------------------------------------------------
// B fragment loader: B = X^T tile (32x16 f16). Lane l16 = column N (== row
// nbase+l16 of X); lanes 0-15 hold K 0-15, lanes 16-31 hold K 16-31 ->
// element e -> k = kt*32 + hi*16 + e, one 32 B contiguous chunk per lane.
// ---------------------------------------------------------------------------
__device__ __forceinline__ void load_btile(const _Float16* __restrict__ Xh,
                                           int nbase, int l16, int hi,
                                           v16h bf[4]) {
  const _Float16* colp = Xh + (size_t)(nbase + l16) * EMB_D + hi * 16;
#pragma unroll
  for (int kt = 0; kt < 4; ++kt)
    bf[kt] = *(const v16h*)(colp + kt * 32);
}

// ---------------------------------------------------------------------------
// One 16x16 output tile: 4 chained WMMAs (K=128) + fused exp epilogue.
// Diagonal-pair masking + D_pos capture only on the (scalar) diagonal tile.
// ---------------------------------------------------------------------------
__device__ __forceinline__ void compute_tile(const v16h afrag[4],
                                             const v16h bf[4],
                                             int mbase, int mbase_s, int nbase,
                                             int l16, int hi,
                                             float* __restrict__ d_pos,
                                             float acc[8]) {
  v8f c = {};
#pragma unroll
  for (int kt = 0; kt < 4; ++kt)
    c = __builtin_amdgcn_wmma_f32_16x16x32_f16(
        false, afrag[kt], false, bf[kt], (short)0, c, false, false);

  if (nbase == mbase_s) {
    // Diagonal tile (taken once per wave): mask 2x2 pair block, grab D_pos.
    const int col = nbase + l16;
#pragma unroll
    for (int r = 0; r < 8; ++r) {
      const int row = mbase + hi * 8 + r;
      if (((row & 1) == 0) && (col == row + 1))
        d_pos[row >> 1] = c[r] * (1.0f / (float)EMB_D);
      const bool excl = ((row >> 1) == (col >> 1));
      acc[r] += excl ? 0.0f : __builtin_amdgcn_exp2f(fmaf(c[r], K1_EXP, K2_EXP));
    }
  } else {
    // Hot path: 3 VALU per element (fma, exp2, add).
#pragma unroll
    for (int r = 0; r < 8; ++r)
      acc[r] += __builtin_amdgcn_exp2f(fmaf(c[r], K1_EXP, K2_EXP));
  }
}

// ---------------------------------------------------------------------------
// Kernel 2: fused X*X^T/d -> exp(+margin) -> masked row sums + D_pos.
// 64 blocks x 8 waves; wave owns 16 rows (A panel resident in VGPRs),
// sweeps all 512 column tiles with double-buffered B fragments.
// ---------------------------------------------------------------------------
__global__ void __launch_bounds__(256)
fused_simloss_rowsum(const _Float16* __restrict__ Xh,
                     float* __restrict__ row_sum,
                     float* __restrict__ d_pos) {
  const int tid  = threadIdx.x;
  const int wave = tid >> 5;
  const int lane = tid & 31;
  const int l16  = lane & 15;
  const int hi   = lane >> 4;

  const int mbase   = blockIdx.x * 128 + wave * 16;
  const int mbase_s = __builtin_amdgcn_readfirstlane(mbase);  // scalar copy

  // ---- A fragments (rows mbase..mbase+15, all K=128) resident in VGPRs ----
  // A 16x32 f16, this code's k order: lane l16 = row M; hi half selects K 0-15 vs 16-31:
  // element e -> k = kt*32 + hi*16 + e (matches B fragment order).
  v16h afrag[4];
  {
    const _Float16* rowp = Xh + (size_t)(mbase + l16) * EMB_D;
#pragma unroll
    for (int kt = 0; kt < 4; ++kt) {
      const int kb = kt * 32 + hi * 16;
      afrag[kt] = *(const v16h*)(rowp + kb);
    }
  }

  float acc[8];
#pragma unroll
  for (int r = 0; r < 8; ++r) acc[r] = 0.0f;

  // ---- double-buffered sweep over all 512 column tiles --------------------
  v16h b0[4], b1[4];
  load_btile(Xh, 0, l16, hi, b0);
  for (int nt = 0; nt < N_TILES; nt += 2) {
    load_btile(Xh, (nt + 1) * 16, l16, hi, b1);            // prefetch odd tile
    compute_tile(afrag, b0, mbase, mbase_s, nt * 16, l16, hi, d_pos, acc);
    if (nt + 2 < N_TILES)
      load_btile(Xh, (nt + 2) * 16, l16, hi, b0);          // prefetch even tile
    compute_tile(afrag, b1, mbase, mbase_s, (nt + 1) * 16, l16, hi, d_pos, acc);
  }

  // ---- butterfly-reduce row sums across the 16 lanes of each half ---------
#pragma unroll
  for (int off = 1; off < 16; off <<= 1) {
#pragma unroll
    for (int r = 0; r < 8; ++r)
      acc[r] += __shfl_xor(acc[r], off, 32);
  }
  if (l16 == 0) {
#pragma unroll
    for (int r = 0; r < 8; ++r)
      row_sum[mbase + hi * 8 + r] = acc[r];
  }
}

// ---------------------------------------------------------------------------
// Kernel 3: per-pair loss + reduction.
// ---------------------------------------------------------------------------
__global__ void __launch_bounds__(256)
finalize_loss(const float* __restrict__ row_sum,
              const float* __restrict__ d_pos,
              float* __restrict__ out) {
  __shared__ float sdata[256];
  float local = 0.0f;
  for (int p = threadIdx.x; p < N_PAIRS; p += 256) {
    const float S = row_sum[2 * p] + row_sum[2 * p + 1];
    const float J = __logf(EPS_LOSS + S) - d_pos[p];
    const float m = fmaxf(J, 0.0f);
    local += m * m;
  }
  sdata[threadIdx.x] = local;
  __syncthreads();
  for (int s = 128; s > 0; s >>= 1) {
    if (threadIdx.x < s) sdata[threadIdx.x] += sdata[threadIdx.x + s];
    __syncthreads();
  }
  if (threadIdx.x == 0)
    out[0] = sdata[0] / (2.0f * (float)N_PAIRS);
}

// ---------------------------------------------------------------------------
extern "C" void kernel_launch(void* const* d_in, const int* in_sizes, int n_in,
                              void* d_out, int out_size, void* d_ws, size_t ws_size,
                              hipStream_t stream) {
  const float* X = (const float*)d_in[0];
  float* out = (float*)d_out;

  // Workspace: f16 X copy (2 MB) | row_sum f32 (32 KB) | D_pos f32 (16 KB)
  _Float16* Xh   = (_Float16*)d_ws;
  float* row_sum = (float*)((char*)d_ws + (size_t)B_ROWS * EMB_D * sizeof(_Float16));
  float* d_pos   = row_sum + B_ROWS;

  const int n_elems = B_ROWS * EMB_D;
  cvt_f32_to_f16<<<(n_elems / 8 + 255) / 256, 256, 0, stream>>>(X, Xh, n_elems);

  fused_simloss_rowsum<<<B_ROWS / 128, 256, 0, stream>>>(Xh, row_sum, d_pos);

  finalize_loss<<<1, 256, 0, stream>>>(row_sum, d_pos, out);
}
